// SoftAttnGRU_83141976916931
// MI455X (gfx1250) — matrix operand order repaired and checked by reading, not compiled, not run on hardware
//
#include <hip/hip_runtime.h>
#include <hip/hip_bf16.h>

// ---------------------------------------------------------------------------
// SoftAttnGRU for MI455X (gfx1250, wave32, WMMA bf16, LDS-staged B operands)
//   Phase 1: x_proj = inputs @ kernel + bias   (65536x512 * 512x1536, WMMA)
//   Phase 2: persistent-grid scan over T=512; R-slice LDS-resident all steps
// ---------------------------------------------------------------------------

typedef __attribute__((ext_vector_type(16))) __bf16 v16bf;
typedef __attribute__((ext_vector_type(8)))  float  v8f;
typedef __attribute__((ext_vector_type(4)))  int    v4i_;

#define B_   128
#define T_   512
#define D_   512
#define U_   512
#define N3_  1536           // 3*U
#define SCAN_BLOCKS 32      // 32 blocks * 16 cols = 512 h-columns

#define BS_STRIDE 72        // phase-1 LDS row stride in bf16 (64 data + 8 pad, 16B mult)
#define RS_STRIDE 56        // phase-2 LDS row stride in bf16 (48 data + 8 pad, 16B mult)

#if __has_builtin(__builtin_amdgcn_global_load_async_to_lds_b128) && \
    __has_builtin(__builtin_amdgcn_s_wait_asynccnt)
#define HAVE_ASYNC_LDS 1
#endif

// copy 16B global -> LDS (async-to-LDS path on gfx1250 when available)
static __device__ inline void cp16_g2l(const __bf16* g, __bf16* l) {
#ifdef HAVE_ASYNC_LDS
    __builtin_amdgcn_global_load_async_to_lds_b128(
        (__attribute__((address_space(1))) v4i_*)(void*)g,
        (__attribute__((address_space(3))) v4i_*)(void*)l, 0, 0);
#else
    *(uint4*)l = *(const uint4*)g;
#endif
}
static __device__ inline void cp_g2l_wait() {
#ifdef HAVE_ASYNC_LDS
    __builtin_amdgcn_s_wait_asynccnt(0);
#endif
}

// 16 contiguous bf16 -> v16bf (two 16B loads; works for global or LDS src)
static __device__ inline v16bf ld16bf(const __bf16* p) {
    union { uint4 u[2]; v16bf v; } c;
    c.u[0] = *(const uint4*)(p);
    c.u[1] = *(const uint4*)(p + 8);
    return c.v;
}

// A-fragment per ISA layout: lane holds row l&15; half = l>>4 selects
// K runs [half*8, +8) and [16+half*8, +8). p points at (row, kbase+half*8).
static __device__ inline v16bf ld_afrag(const __bf16* p) {
    union { uint4 u[2]; v16bf v; } c;
    c.u[0] = *(const uint4*)(p);
    c.u[1] = *(const uint4*)(p + 16);
    return c.v;
}

static __device__ inline float hsig(float x) {
    return fminf(fmaxf(0.2f * x + 0.5f, 0.0f), 1.0f);
}

// ---------------------------------------------------------------------------
// f32 -> bf16 conversion (grid-stride)
// ---------------------------------------------------------------------------
__global__ void cvt_bf16_kernel(const float* __restrict__ s, __bf16* __restrict__ d, size_t n) {
    size_t i  = (size_t)blockIdx.x * blockDim.x + threadIdx.x;
    size_t st = (size_t)gridDim.x * blockDim.x;
    for (; i < n; i += st) d[i] = (__bf16)s[i];
}

// zero h0 (f32 + bf16 mirrors) and the barrier counter — every launch
__global__ void init_state_kernel(float* hb0, __bf16* hx0, unsigned int* ctr) {
    size_t i  = (size_t)blockIdx.x * blockDim.x + threadIdx.x;
    size_t st = (size_t)gridDim.x * blockDim.x;
    for (; i < (size_t)B_ * U_; i += st) { hb0[i] = 0.0f; hx0[i] = (__bf16)0.0f; }
    if (blockIdx.x == 0 && threadIdx.x == 0) *ctr = 0u;
}

// ---------------------------------------------------------------------------
// Phase 1: x_proj GEMM.
// Block = 256 thr (8 waves), block tile = 256 rows x 64 cols.
// W-slice (512K x 64 cols) staged in LDS once.  Single hot kk-loop with
// register blocking: each wave accumulates TWO 16-row tiles x 4 col-tiles,
// so every LDS B-fragment feeds 2 WMMAs and is consumed immediately
// (nothing loop-invariant left for LICM to hoist/spill).
// ---------------------------------------------------------------------------
__global__ __launch_bounds__(256) void xproj_gemm_kernel(
    const __bf16* __restrict__ Abf,   // (B*T, D) bf16 row-major
    const __bf16* __restrict__ Wbf,   // (D, 3U)  bf16 row-major
    const float*  __restrict__ bias,  // (3U,)
    float*        __restrict__ xproj) // (B*T, 3U) f32
{
    __shared__ __bf16 Bs[D_ * BS_STRIDE];          // 72 KB

    const int lane = threadIdx.x & 31;
    const int wave = threadIdx.x >> 5;
    const int half = lane >> 4;
    const int l16  = lane & 15;
    const int colblk = blockIdx.x * 64;
    const int rowbase = blockIdx.y * 256;

    // cooperative stage of W columns [colblk, colblk+64) for all 512 K rows
    for (int r = threadIdx.x; r < D_; r += 256) {
        const __bf16* src = Wbf + (size_t)r * N3_ + colblk;
        __bf16* dst = &Bs[r * BS_STRIDE];
        #pragma unroll
        for (int ch = 0; ch < 8; ++ch)
            cp16_g2l(src + ch * 8, dst + ch * 8);
    }
    cp_g2l_wait();
    __syncthreads();

    const int r0 = rowbase + wave * 16;        // row-tile 0 for this wave
    const int r1 = rowbase + 128 + wave * 16;  // row-tile 1 for this wave
    const __bf16* arow0 = Abf + (size_t)(r0 + l16) * D_ + half * 8;
    const __bf16* arow1 = Abf + (size_t)(r1 + l16) * D_ + half * 8;

    v8f acc0[4] = {v8f{}, v8f{}, v8f{}, v8f{}};
    v8f acc1[4] = {v8f{}, v8f{}, v8f{}, v8f{}};

    v16bf a0 = ld_afrag(arow0);                // software-pipelined A prefetch
    v16bf a1 = ld_afrag(arow1);

    for (int kk = 0; kk < D_; kk += 32) {
        const int kn = (kk + 32) & (D_ - 1);
        v16bf a0n = ld_afrag(arow0 + kn);
        v16bf a1n = ld_afrag(arow1 + kn);
        const __bf16* brow = &Bs[(kk + half * 8) * BS_STRIDE + l16];
        #pragma unroll
        for (int n = 0; n < 4; ++n) {
            v16bf b;
            #pragma unroll
            for (int s = 0; s < 16; ++s)
                b[s] = brow[((s & 7) + (s >> 3) * 16) * BS_STRIDE + n * 16];
            acc0[n] = __builtin_amdgcn_wmma_f32_16x16x32_bf16(
                false, a0, false, b, (short)0, acc0[n], false, false);
            acc1[n] = __builtin_amdgcn_wmma_f32_16x16x32_bf16(
                false, a1, false, b, (short)0, acc1[n], false, false);
        }
        a0 = a0n;
        a1 = a1n;
    }

    #pragma unroll
    for (int n = 0; n < 4; ++n) {
        const int col = colblk + n * 16 + l16;
        const float bs = bias[col];
        #pragma unroll
        for (int j = 0; j < 8; ++j) {          // C layout: M = j + 8*half
            xproj[(size_t)(r0 + j + 8 * half) * N3_ + col] = acc0[n][j] + bs;
            xproj[(size_t)(r1 + j + 8 * half) * N3_ + col] = acc1[n][j] + bs;
        }
    }
}

// ---------------------------------------------------------------------------
// Phase 2: persistent scan.  32 blocks x 256 thr, device barrier per step.
// Block b owns h columns [16b, 16b+16); its R-slice (512 K x 48 gate-cols)
// lives in LDS for the whole scan.  Wave w owns batch rows [16w, 16w+16).
// (The per-step __syncthreads() fences LDS, so the Rs reads stay in-loop.)
// ---------------------------------------------------------------------------
__global__ __launch_bounds__(256) void gru_scan_kernel(
    const float*  __restrict__ xproj,  // (B*T, 3U) f32
    const float*  __restrict__ attn,   // (B, T)
    const __bf16* __restrict__ Rbf,    // (U, 3U) bf16
    float* hb0, float* hb1,            // f32 h ping-pong
    __bf16* hx0, __bf16* hx1,          // bf16 h ping-pong
    unsigned int* ctr,
    float* __restrict__ out)           // (B, U)
{
    __shared__ __bf16 Rs[U_ * RS_STRIDE];          // 56 KB

    const int lane = threadIdx.x & 31;
    const int wave = threadIdx.x >> 5;
    const int half = lane >> 4;
    const int l16  = lane & 15;
    const int u0   = blockIdx.x * 16;  // this block's h-column base
    const int row0 = wave * 16;        // this wave's batch-row base

    // stage Rz|Rr|Rh columns [u0, u0+16) for all 512 K rows — reused 512 steps
    for (int r = threadIdx.x; r < U_; r += 256) {
        const __bf16* src = Rbf + (size_t)r * N3_ + u0;
        __bf16* dst = &Rs[r * RS_STRIDE];
        #pragma unroll
        for (int g = 0; g < 3; ++g) {
            cp16_g2l(src + g * U_,     dst + g * 16);
            cp16_g2l(src + g * U_ + 8, dst + g * 16 + 8);
        }
    }
    cp_g2l_wait();
    __syncthreads();

    for (int t = 0; t < T_; ++t) {
        const float*  hin   = (t & 1) ? hb1 : hb0;
        float*        hout  = (t & 1) ? hb0 : hb1;
        const __bf16* hxin  = (t & 1) ? hx1 : hx0;
        __bf16*       hxout = (t & 1) ? hx0 : hx1;

        v8f za = {}, ra = {}, ha = {};
        const __bf16* arow = hxin + (size_t)(row0 + l16) * U_ + half * 8;

        v16bf a = ld_afrag(arow);                  // prefetch k-chunk 0
        #pragma unroll 4
        for (int kk = 0; kk < U_; kk += 32) {
            v16bf a_n = ld_afrag(arow + ((kk + 32) & (U_ - 1)));
            const __bf16* rrow = &Rs[(kk + half * 8) * RS_STRIDE + l16];
            v16bf bz, br, bh;
            #pragma unroll
            for (int s = 0; s < 16; ++s) {
                const int o = ((s & 7) + (s >> 3) * 16) * RS_STRIDE;
                bz[s] = rrow[o];
                br[s] = rrow[o + 16];
                bh[s] = rrow[o + 32];
            }
            za = __builtin_amdgcn_wmma_f32_16x16x32_bf16(false, a, false, bz, (short)0, za, false, false);
            ra = __builtin_amdgcn_wmma_f32_16x16x32_bf16(false, a, false, br, (short)0, ra, false, false);
            ha = __builtin_amdgcn_wmma_f32_16x16x32_bf16(false, a, false, bh, (short)0, ha, false, false);
            a = a_n;
        }

        // elementwise gates; C layout: M = j + 8*half, N = l16
        #pragma unroll
        for (int j = 0; j < 8; ++j) {
            const int b = row0 + j + 8 * half;    // batch index
            const int c = u0 + l16;               // hidden unit
            const size_t xbase = ((size_t)b * T_ + t) * N3_ + c;
            const float xz = xproj[xbase];
            const float xr = xproj[xbase + U_];
            const float xh = xproj[xbase + 2 * U_];
            const float ag = attn[(size_t)b * T_ + t];
            const float hp = hin[(size_t)b * U_ + c];

            const float z  = hsig(xz + za[j]);
            const float rg = hsig(xr + ra[j]);
            const float hh = tanhf(xh + rg * ha[j]);
            float hn = z * hp + (1.0f - z) * hh;
            hn = ag * hn + (1.0f - ag) * hp;

            hout[(size_t)b * U_ + c]  = hn;
            hxout[(size_t)b * U_ + c] = (__bf16)hn;
            if (t == T_ - 1) out[(size_t)b * U_ + c] = hn;
        }

        // ---- device-wide barrier between steps (monotonic counter) ----
        __threadfence();
        __syncthreads();
        if (threadIdx.x == 0) {
            __hip_atomic_fetch_add(ctr, 1u, __ATOMIC_RELEASE, __HIP_MEMORY_SCOPE_AGENT);
            const unsigned int target = (unsigned int)SCAN_BLOCKS * (unsigned int)(t + 1);
            while (__hip_atomic_load(ctr, __ATOMIC_ACQUIRE, __HIP_MEMORY_SCOPE_AGENT) < target) {
                __builtin_amdgcn_s_sleep(1);
            }
        }
        __syncthreads();
        __threadfence();
    }
}

// ---------------------------------------------------------------------------
// Launch
// ---------------------------------------------------------------------------
extern "C" void kernel_launch(void* const* d_in, const int* in_sizes, int n_in,
                              void* d_out, int out_size, void* d_ws, size_t ws_size,
                              hipStream_t stream) {
    (void)in_sizes; (void)n_in; (void)out_size; (void)ws_size;
    const float* inputs = (const float*)d_in[0];   // (B,T,D)
    const float* attn   = (const float*)d_in[1];   // (B,T,1)
    const float* kern   = (const float*)d_in[2];   // (D,3U)
    const float* rker   = (const float*)d_in[3];   // (U,3U)
    const float* bias   = (const float*)d_in[4];   // (3U,)
    float* out = (float*)d_out;                    // (B,U)

    // workspace carve-up (256B aligned)
    char* ws = (char*)d_ws;
    size_t off = 0;
    auto take = [&](size_t bytes) { void* p = ws + off; off = (off + bytes + 255) & ~(size_t)255; return p; };
    __bf16* in_bf = (__bf16*)take((size_t)B_ * T_ * D_ * 2);
    __bf16* w_bf  = (__bf16*)take((size_t)D_ * N3_ * 2);
    __bf16* r_bf  = (__bf16*)take((size_t)U_ * N3_ * 2);
    float*  xproj = (float*) take((size_t)B_ * T_ * N3_ * 4);
    float*  hb0   = (float*) take((size_t)B_ * U_ * 4);
    float*  hb1   = (float*) take((size_t)B_ * U_ * 4);
    __bf16* hx0   = (__bf16*)take((size_t)B_ * U_ * 2);
    __bf16* hx1   = (__bf16*)take((size_t)B_ * U_ * 2);
    unsigned int* ctr = (unsigned int*)take(256);

    // convert to bf16
    cvt_bf16_kernel<<<2048, 256, 0, stream>>>(inputs, in_bf, (size_t)B_ * T_ * D_);
    cvt_bf16_kernel<<<512,  256, 0, stream>>>(kern,   w_bf,  (size_t)D_ * N3_);
    cvt_bf16_kernel<<<512,  256, 0, stream>>>(rker,   r_bf,  (size_t)U_ * N3_);
    init_state_kernel<<<64, 256, 0, stream>>>(hb0, hx0, ctr);

    // phase 1: x_proj GEMM  (grid: 24 col-blocks x 256 row-blocks of 256 rows)
    xproj_gemm_kernel<<<dim3(N3_ / 64, (B_ * T_) / 256), 256, 0, stream>>>(
        in_bf, w_bf, bias, xproj);

    // phase 2: persistent scan
    gru_scan_kernel<<<SCAN_BLOCKS, 256, 0, stream>>>(
        xproj, attn, r_bf, hb0, hb1, hx0, hx1, ctr, out);
}
